// MambaEncoder_2405181685826
// MI455X (gfx1250) — matrix operand order repaired and checked by reading, not compiled, not run on hardware
//
#include <hip/hip_runtime.h>
#include <math.h>

// ---------------------------------------------------------------------------
// MambaEncoder for MI455X (gfx1250, wave32, WMMA + Tensor Data Mover).
// Activations are [C, L] channel-major so every channel-mixing einsum is a
// GEMM  Y[M,L] = W[M,K] * X[K,L], run on v_wmma_f32_16x16x32_f16.
// The X tile is DMA'd into LDS by the TDM (tensor_load_to_lds, TENSORcnt),
// then converted f32->f16 into the wave32 fragment layout.
// ---------------------------------------------------------------------------

#define EPS 1e-5f
#define GDIV(a, b) (((a) + (b)-1) / (b))

typedef __attribute__((ext_vector_type(16))) _Float16 v16h;
typedef __attribute__((ext_vector_type(8))) float v8f;
typedef unsigned int u32x4 __attribute__((ext_vector_type(4)));
typedef int i32x8 __attribute__((ext_vector_type(8)));
typedef int i32x4 __attribute__((ext_vector_type(4)));

enum { ACT_NONE = 0, ACT_GELU = 1, ACT_SOFTPLUS = 2 };

#if __has_builtin(__builtin_amdgcn_tensor_load_to_lds) && \
    __has_builtin(__builtin_amdgcn_s_wait_tensorcnt)
#define USE_TDM 1
#else
#define USE_TDM 0
#endif

__device__ __forceinline__ float geluf(float x) {
  // jax.nn.gelu (approximate=True, tanh form)
  float x3 = x * x * x;
  return 0.5f * x * (1.f + tanhf(0.7978845608028654f * (x + 0.044715f * x3)));
}
__device__ __forceinline__ float siluf(float x) { return x / (1.f + expf(-x)); }
__device__ __forceinline__ float softplusf(float x) {
  return fmaxf(x, 0.f) + log1pf(expf(-fabsf(x)));
}

// ---------------------------------------------------------------------------
// WMMA GEMM:  Y[M,N] = act(W[M,K] * X[K,N] + bias[M])
// Block = 128 threads (4 waves). Block tile: 16(M) x 256(N); each wave owns a
// 16x64 slab = 4 accumulators, 4 WMMAs per K-step sharing one A fragment.
// K is zero-padded to 32 (TDM zero-fills OOB; fallback path guards).
// ---------------------------------------------------------------------------
#define TN 256
__global__ void gemm_wmma(const float* __restrict__ W, const float* __restrict__ X,
                          const float* __restrict__ bias, float* __restrict__ Y,
                          int M, int K, int N, int act) {
  __shared__ _Float16 Alds[16 * 32];   // [m][k] f16
  __shared__ _Float16 Blds[TN * 32];   // [n][k] f16 fragment layout
#if USE_TDM
  __shared__ float Bstage[32 * TN];    // [k][n] f32, filled by TDM
#endif
  int tid = threadIdx.x;
  int wave = tid >> 5;
  int lane = tid & 31;
  int mBase = blockIdx.y * 16;
  int nBlock = blockIdx.x * TN;

  v8f acc[4] = {};

  for (int k0 = 0; k0 < K; k0 += 32) {
#if USE_TDM
    // --- Tensor Data Mover: DMA 32 x 256 f32 tile of X into LDS staging. ---
    // D# per CDNA5 ISA. tensor_dim0/1 = remaining
    // N/K so the TDM zero-fills the out-of-bounds part of the tile.
    if (wave == 0) {
      unsigned long long ga =
          (unsigned long long)(uintptr_t)(X + (size_t)k0 * N + nBlock);
      unsigned ldsOff = (unsigned)(uintptr_t)&Bstage[0];  // LDS byte offset
      unsigned td0 = (unsigned)(N - nBlock);              // remaining cols
      unsigned td1 = (unsigned)(K - k0);                  // remaining rows
      unsigned long long st0 = (unsigned long long)N;     // row stride (elems)
      u32x4 g0;
      g0[0] = 1u;                                   // count=1, no gather
      g0[1] = ldsOff;                               // lds_addr
      g0[2] = (unsigned)(ga & 0xFFFFFFFFu);         // global_addr[31:0]
      g0[3] = (unsigned)((ga >> 32) & 0x01FFFFFFu)  // global_addr[56:32]
              | 0x80000000u;                        // type=2 ("image")
      i32x8 g1;
      g1[0] = 0x20000;                              // data_size=2 (4 bytes)
      g1[1] = (int)((td0 & 0xFFFFu) << 16);         // tensor_dim0[15:0]
      g1[2] = (int)((td0 >> 16) | ((td1 & 0xFFFFu) << 16));
      g1[3] = (int)((td1 >> 16) | (256u << 16));    // tile_dim0 = 256
      g1[4] = 32;                                   // tile_dim1=32, tile_dim2=0
      g1[5] = (int)(st0 & 0xFFFFFFFFull);           // dim0_stride[31:0]
      g1[6] = (int)((st0 >> 32) & 0xFFFFull);       // dim0_stride[47:32]
      g1[7] = 0;
      i32x4 gz = {0, 0, 0, 0};
#if __has_include(<hip/amd_detail/amd_gfx1250_TDM.h>)
      i32x8 gz8 = {0, 0, 0, 0, 0, 0, 0, 0};
      __builtin_amdgcn_tensor_load_to_lds(g0, g1, gz, gz, gz8, 0);
#else
      __builtin_amdgcn_tensor_load_to_lds(g0, g1, gz, gz, 0);
#endif
    }
#endif
    // stage A tile (16x32 f16), zero-padded, all threads
    for (int i = tid; i < 16 * 32; i += 128) {
      int m = i >> 5, k = i & 31;
      float v = 0.f;
      if (mBase + m < M && k0 + k < K) v = W[(size_t)(mBase + m) * K + (k0 + k)];
      Alds[m * 32 + k] = (_Float16)v;
    }
#if USE_TDM
    if (wave == 0) __builtin_amdgcn_s_wait_tensorcnt(0);
    __syncthreads();  // staging + A visible to all waves
    // convert staged f32 [k][n] -> f16 [n][k] fragment layout
    for (int i = tid; i < TN * 32; i += 128) {
      int k = i >> 8, n = i & (TN - 1);
      Blds[n * 32 + k] = (_Float16)Bstage[k * TN + n];
    }
#else
    // fallback: direct global -> f16 LDS, coalesced over n
    for (int i = tid; i < TN * 32; i += 128) {
      int k = i >> 8, n = i & (TN - 1);
      float v = 0.f;
      if (nBlock + n < N && k0 + k < K) v = X[(size_t)(k0 + k) * N + (nBlock + n)];
      Blds[n * 32 + k] = (_Float16)v;
    }
    if (k0 + 32 < K) {
      const float* p = &X[(size_t)(k0 + 32) * N + nBlock + tid];
      __builtin_prefetch(p, 0, 1);  // global_prefetch_b8
    }
#endif
    __syncthreads();

    // Fragments per ISA 16-bit A 16x32 / B 32x16 wave32 layouts.
    int mr = lane & 15;
    int hi = lane >> 4;
    v16h afrag;
#pragma unroll
    for (int r = 0; r < 8; ++r) {
      int kk = (r < 4 ? 2 * r : 16 + 2 * (r - 4)) + 8 * hi;
      afrag[2 * r] = Alds[mr * 32 + kk];
      afrag[2 * r + 1] = Alds[mr * 32 + kk + 1];
    }
#pragma unroll
    for (int sub = 0; sub < 4; ++sub) {
      int nloc = wave * 64 + sub * 16 + mr;
      v16h bfrag;
#pragma unroll
      for (int r = 0; r < 16; ++r) bfrag[r] = Blds[nloc * 32 + (r < 8 ? 8 * hi + r : 16 + 8 * hi + (r - 8))];
      acc[sub] = __builtin_amdgcn_wmma_f32_16x16x32_f16(false, afrag, false, bfrag,
                                                        (short)0, acc[sub], false, false);
    }
    __syncthreads();
  }

  // D layout: vgpr r, lanes 0-15 -> row r, lanes 16-31 -> row 8+r; col = lane&15
#pragma unroll
  for (int sub = 0; sub < 4; ++sub) {
    int col = nBlock + wave * 64 + sub * 16 + (lane & 15);
    if (col < N) {
#pragma unroll
      for (int r = 0; r < 8; ++r) {
        int row = mBase + r + ((lane >> 4) * 8);
        if (row < M) {
          float v = acc[sub][r];
          if (bias) v += bias[row];
          if (act == ACT_GELU) v = geluf(v);
          else if (act == ACT_SOFTPLUS) v = softplusf(v);
          Y[(size_t)row * N + col] = v;
        }
      }
    }
  }
}

// ---------------------------------------------------------------------------
// Convolutions
// ---------------------------------------------------------------------------
__global__ void conv_stem(const float* __restrict__ x, const float* __restrict__ w,
                          const float* __restrict__ b, float* __restrict__ y) {
  // 1x64^3 -> 48x32^3, k=7 s=2 p=3
  int i = blockIdx.x * 256 + threadIdx.x;
  if (i >= 48 * 32768) return;
  int c = i >> 15, v = i & 32767;
  int od = v >> 10, oh = (v >> 5) & 31, ow = v & 31;
  float s = b[c];
  const float* wc = w + c * 343;
  for (int kd = 0; kd < 7; kd++) {
    int id = od * 2 - 3 + kd;
    if ((unsigned)id >= 64u) continue;
    for (int kh = 0; kh < 7; kh++) {
      int ih = oh * 2 - 3 + kh;
      if ((unsigned)ih >= 64u) continue;
      for (int kw = 0; kw < 7; kw++) {
        int iw = ow * 2 - 3 + kw;
        if ((unsigned)iw >= 64u) continue;
        s += wc[kd * 49 + kh * 7 + kw] * x[(id * 64 + ih) * 64 + iw];
      }
    }
  }
  y[i] = s;
}

__global__ void conv_down(const float* __restrict__ x, const float* __restrict__ w,
                          const float* __restrict__ b, float* __restrict__ y,
                          int Cin, int Co, int Din, int Dout) {
  int Lout = Dout * Dout * Dout;
  int i = blockIdx.x * 256 + threadIdx.x;
  if (i >= Co * Lout) return;
  int c = i / Lout, v = i % Lout;
  int od = v / (Dout * Dout), oh = (v / Dout) % Dout, ow = v % Dout;
  float s = b[c];
  const float* wc = w + (size_t)c * Cin * 8;
  for (int ci = 0; ci < Cin; ci++) {
    const float* xc = x + (size_t)ci * Din * Din * Din;
    const float* wci = wc + ci * 8;
    for (int kd = 0; kd < 2; kd++)
      for (int kh = 0; kh < 2; kh++)
        for (int kw = 0; kw < 2; kw++)
          s += wci[kd * 4 + kh * 2 + kw] *
               xc[((od * 2 + kd) * Din + oh * 2 + kh) * (size_t)Din + ow * 2 + kw];
  }
  y[i] = s;
}

__global__ void dwconv3(const float* __restrict__ x, const float* __restrict__ w,
                        const float* __restrict__ b, float* __restrict__ y, int C, int Dm) {
  int L = Dm * Dm * Dm;
  int i = blockIdx.x * 256 + threadIdx.x;
  if (i >= C * L) return;
  int c = i / L, v = i % L;
  int od = v / (Dm * Dm), oh = (v / Dm) % Dm, ow = v % Dm;
  float s = b[c];
  const float* wc = w + c * 27;
  const float* xc = x + (size_t)c * L;
  for (int kd = 0; kd < 3; kd++) {
    int id = od - 1 + kd;
    if ((unsigned)id >= (unsigned)Dm) continue;
    for (int kh = 0; kh < 3; kh++) {
      int ih = oh - 1 + kh;
      if ((unsigned)ih >= (unsigned)Dm) continue;
      for (int kw = 0; kw < 3; kw++) {
        int iw = ow - 1 + kw;
        if ((unsigned)iw >= (unsigned)Dm) continue;
        s += wc[kd * 9 + kh * 3 + kw] * xc[(id * Dm + ih) * Dm + iw];
      }
    }
  }
  y[i] = s;
}

// causal depthwise conv1d (4 taps, left pad 3) + SiLU; rev=1 runs anti-causal
__global__ void cconv_silu(const float* __restrict__ x, const float* __restrict__ w,
                           const float* __restrict__ b, float* __restrict__ y,
                           int di, int L, int rev) {
  int i = blockIdx.x * 256 + threadIdx.x;
  if (i >= di * L) return;
  int d = i / L, l = i % L;
  const float* xd = x + (size_t)d * L;
  float s = b[d];
  for (int k = 0; k < 4; k++) {
    int src = rev ? (l + 3 - k) : (l + k - 3);
    if ((unsigned)src < (unsigned)L) s += w[d * 4 + k] * xd[src];
  }
  y[i] = siluf(s);
}

// ---------------------------------------------------------------------------
// Norms
// ---------------------------------------------------------------------------
__global__ void instnorm(const float* __restrict__ x, float* __restrict__ y,
                         const float* __restrict__ gamma, const float* __restrict__ beta,
                         int L) {
  int c = blockIdx.x;
  const float* xc = x + (size_t)c * L;
  float* yc = y + (size_t)c * L;
  __shared__ float rs[256], rq[256];
  float s = 0.f, q = 0.f;
  for (int i = threadIdx.x; i < L; i += 256) {
    float v = xc[i];
    s += v;
    q += v * v;
  }
  rs[threadIdx.x] = s;
  rq[threadIdx.x] = q;
  __syncthreads();
  for (int o = 128; o; o >>= 1) {
    if (threadIdx.x < o) {
      rs[threadIdx.x] += rs[threadIdx.x + o];
      rq[threadIdx.x] += rq[threadIdx.x + o];
    }
    __syncthreads();
  }
  float mean = rs[0] / L;
  float var = rq[0] / L - mean * mean;
  float inv = rsqrtf(var + EPS);
  float g = gamma ? gamma[c] : 1.f;
  float bb = beta ? beta[c] : 0.f;
  for (int i = threadIdx.x; i < L; i += 256) yc[i] = (xc[i] - mean) * inv * g + bb;
}

// LayerNorm across channels for [C,L] storage (per spatial position l)
__global__ void layernorm_cl(const float* __restrict__ x, float* __restrict__ y,
                             const float* __restrict__ w, const float* __restrict__ b,
                             int C, int L) {
  int l = blockIdx.x * 256 + threadIdx.x;
  if (l >= L) return;
  float s = 0.f, q = 0.f;
  for (int c = 0; c < C; c++) {
    float v = x[(size_t)c * L + l];
    s += v;
    q += v * v;
  }
  float m = s / C;
  float var = q / C - m * m;
  float inv = rsqrtf(var + EPS);
  for (int c = 0; c < C; c++)
    y[(size_t)c * L + l] = (x[(size_t)c * L + l] - m) * inv * w[c] + b[c];
}

// ---------------------------------------------------------------------------
// Selective scan: wave32, 2 channels per wave, 16 lanes = 16 states/channel.
// h[n] <- exp(delta*A[n])*h[n] + delta*B[n]*u ; y = sum_n h[n]*C[n] + D*u
// ---------------------------------------------------------------------------
__global__ void selective_scan(const float* __restrict__ u, const float* __restrict__ delta,
                               const float* __restrict__ Alog, const float* __restrict__ Bm,
                               const float* __restrict__ Cm, const float* __restrict__ Dp,
                               float* __restrict__ y, int di, int L, int rev) {
  int lane = threadIdx.x & 31;
  int d = blockIdx.x * 2 + (lane >> 4);
  int n = lane & 15;
  float A = -expf(Alog[d * 16 + n]);
  float Dd = Dp[d];
  float h = 0.f;
  const float* ud = u + (size_t)d * L;
  const float* dd = delta + (size_t)d * L;
  const float* Bn = Bm + (size_t)n * L;
  const float* Cn = Cm + (size_t)n * L;
  float* yd = y + (size_t)d * L;
  for (int t = 0; t < L; ++t) {
    int l = rev ? (L - 1 - t) : t;
    float dl = dd[l];
    float ul = ud[l];
    float a = expf(dl * A);
    h = a * h + dl * Bn[l] * ul;
    float p = h * Cn[l];
    p += __shfl_xor(p, 1, 32);
    p += __shfl_xor(p, 2, 32);
    p += __shfl_xor(p, 4, 32);
    p += __shfl_xor(p, 8, 32);
    if (n == 0) yd[l] = p + Dd * ul;
  }
}

// ---------------------------------------------------------------------------
// Elementwise helpers
// ---------------------------------------------------------------------------
__global__ void add2_k(const float* a, const float* b, float* o, int n) {
  int i = blockIdx.x * 256 + threadIdx.x;
  if (i < n) o[i] = a[i] + b[i];
}
__global__ void add3_k(const float* a, const float* b, const float* c, float* o, int n) {
  int i = blockIdx.x * 256 + threadIdx.x;
  if (i < n) o[i] = a[i] + b[i] + c[i];
}
__global__ void addip_k(float* dst, const float* src, int n) {
  int i = blockIdx.x * 256 + threadIdx.x;
  if (i < n) dst[i] += src[i];
}
__global__ void gate_silu_k(const float* a, const float* z, float* o, int n) {
  int i = blockIdx.x * 256 + threadIdx.x;
  if (i < n) o[i] = a[i] * siluf(z[i]);
}
// mode 0: scan-order gather xs[l]=x[(l%nsl)*M + l/nsl]; mode 1: inverse
__global__ void permute_seq(const float* x, float* y, int di, int L, int nsl, int inv) {
  int i = blockIdx.x * 256 + threadIdx.x;
  if (i >= di * L) return;
  int d = i / L, l = i % L, M = L / nsl;
  int src = inv ? ((l % M) * nsl + l / M) : ((l % nsl) * M + l / nsl);
  y[(size_t)d * L + l] = x[(size_t)d * L + src];
}

// ---------------------------------------------------------------------------
// Host orchestration
// ---------------------------------------------------------------------------
extern "C" void kernel_launch(void* const* d_in, const int* in_sizes, int n_in,
                              void* d_out, int out_size, void* d_ws, size_t ws_size,
                              hipStream_t stream) {
  (void)in_sizes;
  (void)out_size;
  (void)ws_size;
  if (n_in < 353) return;  // x + 352 param leaves expected

  const int DIMS[4] = {48, 96, 192, 384};
  const int SPD[4] = {32, 16, 8, 4};
  const int NSL[4] = {64, 32, 16, 8};

  const float* x_in = (const float*)d_in[0];
  int idx = 1;
  auto nx = [&]() -> const float* { return (const float*)d_in[idx++]; };

  struct Branch { const float *A_log, *D, *conv_b, *conv_w, *dt_b, *dt_w, *x_proj; };
  struct MLayer { Branch br_b, br_f, br_s; const float *in_proj, *ln_b, *ln_w, *out_proj; };
  struct GscP { const float *c_b, *c_w, *dw_b, *dw_w, *e_b, *e_w, *gn_b, *gn_w; };
  struct StageP {
    const float *ds_b, *ds_w;
    GscP g[4];
    MLayer m[2];
    const float *fc1_b, *fc1_w, *fc2_b, *fc2_w;
  };
  StageP st[4];
  auto rdBranch = [&](Branch& b) {
    b.A_log = nx(); b.D = nx(); b.conv_b = nx(); b.conv_w = nx();
    b.dt_b = nx(); b.dt_w = nx(); b.x_proj = nx();
  };
  for (int s = 0; s < 4; s++) {
    StageP& p = st[s];
    p.ds_b = nx(); p.ds_w = nx();
    for (int gi = 0; gi < 4; gi++) {
      GscP& g = p.g[gi];
      g.c_b = nx(); g.c_w = nx(); g.dw_b = nx(); g.dw_w = nx();
      g.e_b = nx(); g.e_w = nx(); g.gn_b = nx(); g.gn_w = nx();
    }
    for (int mi = 0; mi < 2; mi++) {
      MLayer& m = p.m[mi];
      rdBranch(m.br_b);
      rdBranch(m.br_f);
      m.in_proj = nx(); m.ln_b = nx(); m.ln_w = nx(); m.out_proj = nx();
      rdBranch(m.br_s);
    }
    p.fc1_b = nx(); p.fc1_w = nx(); p.fc2_b = nx(); p.fc2_w = nx();
  }

  // Workspace arena (floats), sized for stage-0 maxima.
  const size_t CL = 1572864;   // C*L max  (48*32768)
  const size_t C2L = 3145728;  // 2C*L max (di*L)
  const size_t C4L = 6291456;  // 4C*L max (xz)
  float* w = (float*)d_ws;
  float* bx = w;   w += CL;
  float* bx2 = w;  w += CL;
  float* g1 = w;   w += CL;
  float* g2 = w;   w += CL;
  float* g3 = w;   w += CL;
  float* tsm = w;  w += CL;   // x_proj output (dtr+32)*L
  float* t0 = w;   w += C4L;  // xz
  float* t1 = w;   w += C2L;
  float* t2 = w;   w += C2L;
  float* t3 = w;   w += C2L;
  float* t4 = w;   w += C2L;
  float* tyacc = w;

  auto gemmL = [&](const float* W, const float* X, const float* bias, float* Y,
                   int M, int K, int N, int act) {
    dim3 grid(GDIV(N, TN), GDIV(M, 16));
    gemm_wmma<<<grid, dim3(128), 0, stream>>>(W, X, bias, Y, M, K, N, act);
  };
  auto ew = [&](int n) { return dim3(GDIV(n, 256)); };

  auto runMednext = [&](const GscP& g, const float* in, float* out, int C, int Dm) {
    int L = Dm * Dm * Dm, n = C * L;
    dwconv3<<<ew(n), 256, 0, stream>>>(in, g.dw_w, g.dw_b, t1, C, Dm);
    instnorm<<<C, 256, 0, stream>>>(t1, t2, g.gn_w, g.gn_b, L);
    gemmL(g.e_w, t2, g.e_b, t3, 2 * C, C, L, ACT_GELU);
    gemmL(g.c_w, t3, g.c_b, t4, C, 2 * C, L, ACT_NONE);
    add2_k<<<ew(n), 256, 0, stream>>>(in, t4, out, n);
  };

  auto runBranch = [&](const Branch& bp, const float* xin, float* yout, int rev,
                       int di, int L, int dtr) {
    int n = di * L;
    cconv_silu<<<ew(n), 256, 0, stream>>>(xin, bp.conv_w, bp.conv_b, t1, di, L, rev);
    int E = dtr + 32;
    gemmL(bp.x_proj, t1, nullptr, tsm, E, di, L, ACT_NONE);
    gemmL(bp.dt_w, tsm, bp.dt_b, t2, di, dtr, L, ACT_SOFTPLUS);
    selective_scan<<<di / 2, 32, 0, stream>>>(t1, t2, bp.A_log,
                                              tsm + (size_t)dtr * L,
                                              tsm + (size_t)(dtr + 16) * L,
                                              bp.D, yout, di, L, rev);
  };

  float* outp = (float*)d_out;
  size_t ooff = 0;

  for (int s = 0; s < 4; s++) {
    int C = DIMS[s], Dm = SPD[s];
    int L = Dm * Dm * Dm;
    int di = 2 * C;
    int dtr = (C + 15) / 16;

    // ---- downsample ----
    if (s == 0) {
      conv_stem<<<ew(48 * 32768), 256, 0, stream>>>(x_in, st[0].ds_w, st[0].ds_b, bx);
    } else {
      int Cp = DIMS[s - 1], Dp = SPD[s - 1], Lp = Dp * Dp * Dp;
      instnorm<<<Cp, 256, 0, stream>>>(bx, t1, nullptr, nullptr, Lp);
      conv_down<<<ew(C * L), 256, 0, stream>>>(t1, st[s].ds_w, st[s].ds_b, bx2,
                                               Cp, C, Dp, Dm);
      float* tmp = bx; bx = bx2; bx2 = tmp;
    }

    // ---- GSC ----
    int nCL = C * L;
    runMednext(st[s].g[0], bx, g1, C, Dm);
    runMednext(st[s].g[1], g1, g2, C, Dm);
    add2_k<<<ew(nCL), 256, 0, stream>>>(g1, g2, g2, nCL);
    runMednext(st[s].g[2], g2, g3, C, Dm);
    runMednext(st[s].g[3], g3, g1, C, Dm);
    add3_k<<<ew(nCL), 256, 0, stream>>>(g2, g3, g1, bx, nCL);

    // ---- Mamba layers ----
    for (int mi = 0; mi < 2; mi++) {
      const MLayer& m = st[s].m[mi];
      int ndl = di * L;
      layernorm_cl<<<ew(L), 256, 0, stream>>>(bx, t1, m.ln_w, m.ln_b, C, L);
      gemmL(m.in_proj, t1, nullptr, t0, 4 * C, C, L, ACT_NONE);
      float* xm = t0;
      float* z = t0 + (size_t)di * L;

      runBranch(m.br_f, xm, tyacc, 0, di, L, dtr);
      runBranch(m.br_b, xm, t4, 1, di, L, dtr);
      addip_k<<<ew(ndl), 256, 0, stream>>>(tyacc, t4, ndl);
      permute_seq<<<ew(ndl), 256, 0, stream>>>(xm, t4, di, L, NSL[s], 0);
      runBranch(m.br_s, t4, t3, 0, di, L, dtr);
      permute_seq<<<ew(ndl), 256, 0, stream>>>(t3, t4, di, L, NSL[s], 1);
      addip_k<<<ew(ndl), 256, 0, stream>>>(tyacc, t4, ndl);

      gate_silu_k<<<ew(ndl), 256, 0, stream>>>(tyacc, z, t1, ndl);
      gemmL(m.out_proj, t1, nullptr, t2, C, di, L, ACT_NONE);
      addip_k<<<ew(nCL), 256, 0, stream>>>(bx, t2, nCL);
    }

    // ---- MLP head -> output ----
    instnorm<<<C, 256, 0, stream>>>(bx, t1, nullptr, nullptr, L);
    gemmL(st[s].fc1_w, t1, st[s].fc1_b, t3, 2 * C, C, L, ACT_GELU);
    gemmL(st[s].fc2_w, t3, st[s].fc2_b, outp + ooff, C, 2 * C, L, ACT_NONE);
    ooff += (size_t)C * L;
  }
}
